// Block_64922725646971
// MI455X (gfx1250) — matrix operand order repaired and checked by reading, not compiled, not run on hardware
//
#include <hip/hip_runtime.h>
#include <math.h>

// ---------------------------------------------------------------------------
// CDNA5 / gfx1250 WMMA transformer block.  bf16 storage, f32 accumulate.
// Software pipelining via statically-named double buffers (no dynamic
// indexing -> fragments stay in VGPRs, no scratch/LDS spills).
// ---------------------------------------------------------------------------

typedef __attribute__((ext_vector_type(16))) __bf16 v16bf;
typedef __attribute__((ext_vector_type(8)))  float  v8f;

union Frag {
    v16bf    v;
    uint4    q[2];
    unsigned u[8];
};

__device__ __forceinline__ unsigned short f2bf_bits(float f) {
    unsigned u = __float_as_uint(f);
    unsigned r = u + 0x7FFFu + ((u >> 16) & 1u);   // round-to-nearest-even
    return (unsigned short)(r >> 16);
}

__device__ __forceinline__ v8f wmma_bf16(const Frag& a, const Frag& b, v8f c) {
    return __builtin_amdgcn_wmma_f32_16x16x32_bf16(false, a.v, false, b.v,
                                                   (short)0, c, false, false);
}

// A-fragment (16x32 bf16): two 16B chunks at +0 and +16 elements.
__device__ __forceinline__ void load_afrag(Frag& a, const unsigned short* p) {
    a.q[0] = *(const uint4*)(p);
    a.q[1] = *(const uint4*)(p + 16);
}
// B-fragment (32x16 bf16 via Bt[N,K]): two 16B chunks at +0 and +16 elements.
__device__ __forceinline__ void load_bfrag(Frag& b, const unsigned short* p) {
    b.q[0] = *(const uint4*)(p);
    b.q[1] = *(const uint4*)(p + 16);
}

// ---------------------------------------------------------------------------
// Weight cast + transpose:  W[K,N] f32  ->  Wt[N,K] bf16   (K,N mult of 32)
// ---------------------------------------------------------------------------
__global__ __launch_bounds__(256)
void wcastT(const float* __restrict__ W, unsigned short* __restrict__ Wt,
            int K, int N) {
    __shared__ float tile[32][33];
    int nb = blockIdx.x * 32, kb = blockIdx.y * 32;
    int tx = threadIdx.x, ty = threadIdx.y;
#pragma unroll
    for (int i = 0; i < 4; ++i)
        tile[ty + 8 * i][tx] = W[(size_t)(kb + ty + 8 * i) * N + nb + tx];
    __syncthreads();
#pragma unroll
    for (int i = 0; i < 4; ++i)
        Wt[(size_t)(nb + ty + 8 * i) * K + kb + tx] = f2bf_bits(tile[tx][ty + 8 * i]);
}

// ---------------------------------------------------------------------------
// LayerNorm (f32 in) -> bf16 out.  One block per row.
// ---------------------------------------------------------------------------
__global__ __launch_bounds__(256)
void ln_bf16(const float* __restrict__ x, const float* __restrict__ g,
             const float* __restrict__ b, unsigned short* __restrict__ out, int C) {
    int row = blockIdx.x;
    const float* xr = x + (size_t)row * C;
    float s = 0.f, sq = 0.f;
    for (int j = threadIdx.x; j < C; j += 256) { float v = xr[j]; s += v; sq += v * v; }
#pragma unroll
    for (int o = 16; o > 0; o >>= 1) { s += __shfl_xor(s, o); sq += __shfl_xor(sq, o); }
    __shared__ float ss[8], ssq[8];
    int wave = threadIdx.x >> 5, lane = threadIdx.x & 31;
    if (lane == 0) { ss[wave] = s; ssq[wave] = sq; }
    __syncthreads();
    s = 0.f; sq = 0.f;
#pragma unroll
    for (int w = 0; w < 8; ++w) { s += ss[w]; sq += ssq[w]; }
    float mu   = s / (float)C;
    float var  = sq / (float)C - mu * mu;
    float rstd = rsqrtf(var + 1e-5f);
    for (int j = threadIdx.x; j < C; j += 256) {
        float v = (xr[j] - mu) * rstd * g[j] + b[j];
        out[(size_t)row * C + j] = f2bf_bits(v);
    }
}

// ---------------------------------------------------------------------------
// GEMM: Out[M,N] = A[M,K](bf16) * Wt[N,K]^T(bf16) + bias, with epilogues.
//  mode 0: qkv split -> oq[T,C], ok[T,C], ovt[C,T]  (v transposed, all bf16)
//  mode 1: outf = acc + bias + res                   (f32)
//  mode 2: outbf = gelu(acc + bias)                  (bf16, exact gelu)
// Block: 256 thr = 8 waves; block tile 128(M) x 64(N); wave tile 16 x 64.
// K must be a multiple of 64 (call sites: 768 / 3072).
// ---------------------------------------------------------------------------
__global__ __launch_bounds__(256)
void gemm_bf16_wmma(const unsigned short* __restrict__ A,
                    const unsigned short* __restrict__ Bt,
                    const float* __restrict__ bias,
                    int M, int N, int K, int mode,
                    const float* __restrict__ res,
                    float* __restrict__ outf,
                    unsigned short* __restrict__ outbf,
                    unsigned short* __restrict__ oq,
                    unsigned short* __restrict__ ok,
                    unsigned short* __restrict__ ovt,
                    int Cdim, int Tdim) {
    int lane = threadIdx.x & 31;
    int wave = threadIdx.x >> 5;
    int lh = lane & 15, hi = lane >> 4;
    int m0 = blockIdx.y * 128 + wave * 16;
    int n0 = blockIdx.x * 64;

    v8f acc[4];
#pragma unroll
    for (int t = 0; t < 4; ++t)
#pragma unroll
        for (int i = 0; i < 8; ++i) acc[t][i] = 0.0f;

    const unsigned short* ap  = A + (size_t)(m0 + lh) * K + hi * 8;
    const unsigned short* bpb = Bt + (size_t)(n0 + lh) * K + hi * 8;
    const size_t bstride = (size_t)16 * K;          // 16 columns of Bt

    // Two statically-named fragment sets; two K-steps per iteration.
    Frag a0, a1, b0[4], b1[4];
    load_afrag(a0, ap);
#pragma unroll
    for (int t = 0; t < 4; ++t) load_bfrag(b0[t], bpb + t * bstride);

    for (int k0 = 0; k0 < K; k0 += 64) {
        // preload k0+32 while multiplying k0
        load_afrag(a1, ap + k0 + 32);
#pragma unroll
        for (int t = 0; t < 4; ++t) load_bfrag(b1[t], bpb + t * bstride + k0 + 32);
#pragma unroll
        for (int t = 0; t < 4; ++t) acc[t] = wmma_bf16(a0, b0[t], acc[t]);

        // preload k0+64 while multiplying k0+32
        if (k0 + 64 < K) {
            load_afrag(a0, ap + k0 + 64);
#pragma unroll
            for (int t = 0; t < 4; ++t) load_bfrag(b0[t], bpb + t * bstride + k0 + 64);
            __builtin_prefetch(ap + k0 + 96, 0, 1);
        }
#pragma unroll
        for (int t = 0; t < 4; ++t) acc[t] = wmma_bf16(a1, b1[t], acc[t]);
    }

#pragma unroll
    for (int t = 0; t < 4; ++t) {
        int col = n0 + t * 16 + lh;
        float bv = bias[col];
#pragma unroll
        for (int r = 0; r < 8; ++r) {
            int row = m0 + r + hi * 8;
            float v = acc[t][r] + bv;
            if (mode == 0) {
                if (col < Cdim)
                    oq[(size_t)row * Cdim + col] = f2bf_bits(v);
                else if (col < 2 * Cdim)
                    ok[(size_t)row * Cdim + (col - Cdim)] = f2bf_bits(v);
                else
                    ovt[(size_t)(col - 2 * Cdim) * Tdim + row] = f2bf_bits(v);
            } else if (mode == 1) {
                size_t idx = (size_t)row * N + col;
                outf[idx] = v + res[idx];
            } else {
                float gl = 0.5f * v * (1.0f + erff(v * 0.70710678118654752f));
                outbf[(size_t)row * N + col] = f2bf_bits(gl);
            }
        }
    }
}

// ---------------------------------------------------------------------------
// Flash attention (no mask). One wave = one (head, 16-query tile).
// S^T = K·Q^T keeps queries in lanes -> 1-shuffle softmax; exp'd P^T packs
// lane-locally into the A-fragment of Y += P·V (V read from Vt[d,T]).
// Two 32-key sub-chunks per iteration with named K-fragment sets (static
// double buffering); V fragments issued before the softmax VALU work.
// T must be a multiple of 64.
// ---------------------------------------------------------------------------
__global__ __launch_bounds__(256)
void attn_wmma(const unsigned short* __restrict__ Q,
               const unsigned short* __restrict__ Kmat,
               const unsigned short* __restrict__ Vt,
               unsigned short* __restrict__ Y,
               int T, int C, int H) {
    int lane = threadIdx.x & 31, wave = threadIdx.x >> 5;
    int lh = lane & 15, hi = lane >> 4;
    int nq = T >> 4;
    int tile = blockIdx.x * 8 + wave;
    int h = tile / nq, qt = tile - h * nq;
    int m0 = qt << 4;
    int dbase = h * 64;

    Frag bq[2];                                    // Q as B-fragments (d chunks)
#pragma unroll
    for (int dc = 0; dc < 2; ++dc)
        load_bfrag(bq[dc], Q + (size_t)(m0 + lh) * C + dbase + dc * 32 + hi * 8);

    v8f o[4];
#pragma unroll
    for (int t = 0; t < 4; ++t)
#pragma unroll
        for (int i = 0; i < 8; ++i) o[t][i] = 0.0f;
    float m_run = -3.0e38f, l_run = 0.0f;

    const unsigned short* kbase = Kmat + (size_t)lh * C + dbase + hi * 8;
    const unsigned short* vbase = Vt + (size_t)(dbase + lh) * T + hi * 8;

    auto load_akset = [&](Frag (&akc)[4], int kc) {
#pragma unroll
        for (int kt = 0; kt < 2; ++kt)
#pragma unroll
            for (int dc = 0; dc < 2; ++dc)
                load_afrag(akc[kt * 2 + dc],
                           kbase + (size_t)(kc + kt * 16) * C + dc * 32);
    };

    auto process = [&](Frag (&akc)[4], int kc) {
        v8f s1, s2;
#pragma unroll
        for (int i = 0; i < 8; ++i) { s1[i] = 0.0f; s2[i] = 0.0f; }
        s1 = wmma_bf16(akc[0], bq[0], s1);
        s1 = wmma_bf16(akc[1], bq[1], s1);
        s2 = wmma_bf16(akc[2], bq[0], s2);
        s2 = wmma_bf16(akc[3], bq[1], s2);

        Frag bv0, bv1, bv2, bv3;                    // V fragments: issue early
        load_bfrag(bv0, vbase + (size_t)0  * T + kc);
        load_bfrag(bv1, vbase + (size_t)16 * T + kc);
        load_bfrag(bv2, vbase + (size_t)32 * T + kc);
        load_bfrag(bv3, vbase + (size_t)48 * T + kc);

        float mloc = -3.0e38f;
#pragma unroll
        for (int i = 0; i < 8; ++i) {
            s1[i] *= 0.125f; s2[i] *= 0.125f;       // 1/sqrt(64)
            mloc = fmaxf(mloc, fmaxf(s1[i], s2[i]));
        }
        mloc = fmaxf(mloc, __shfl_xor(mloc, 16));
        float mnew  = fmaxf(m_run, mloc);
        float alpha = __expf(m_run - mnew);
        float p1[8], p2[8], psum = 0.0f;
#pragma unroll
        for (int i = 0; i < 8; ++i) {
            p1[i] = __expf(s1[i] - mnew);
            p2[i] = __expf(s2[i] - mnew);
            psum += p1[i] + p2[i];
        }
        l_run = l_run * alpha + psum + __shfl_xor(psum, 16);
        m_run = mnew;

        Frag apf;                                   // P^T -> A-fragment, lane-local
#pragma unroll
        for (int j = 0; j < 4; ++j) {
            apf.u[j]     = (unsigned)f2bf_bits(p1[2 * j]) | ((unsigned)f2bf_bits(p1[2 * j + 1]) << 16);
            apf.u[j + 4] = (unsigned)f2bf_bits(p2[2 * j]) | ((unsigned)f2bf_bits(p2[2 * j + 1]) << 16);
        }
#pragma unroll
        for (int r = 0; r < 8; ++r) {               // per-row rescale of O
            float ar = __shfl(alpha, r + hi * 8);
#pragma unroll
            for (int dt = 0; dt < 4; ++dt) o[dt][r] *= ar;
        }
        o[0] = wmma_bf16(apf, bv0, o[0]);
        o[1] = wmma_bf16(apf, bv1, o[1]);
        o[2] = wmma_bf16(apf, bv2, o[2]);
        o[3] = wmma_bf16(apf, bv3, o[3]);
    };

    Frag ak0[4], ak1[4];                            // static double buffer
    load_akset(ak0, 0);
    for (int kc = 0; kc < T; kc += 64) {
        load_akset(ak1, kc + 32);                   // preload sub-chunk 2
        process(ak0, kc);
        if (kc + 64 < T) load_akset(ak0, kc + 64);  // preload next iteration
        process(ak1, kc + 32);
    }

#pragma unroll
    for (int r = 0; r < 8; ++r) {
        float linv = 1.0f / __shfl(l_run, r + hi * 8);
        int row = m0 + r + hi * 8;
#pragma unroll
        for (int dt = 0; dt < 4; ++dt)
            Y[(size_t)row * C + dbase + dt * 16 + lh] = f2bf_bits(o[dt][r] * linv);
    }
}

// ---------------------------------------------------------------------------
extern "C" void kernel_launch(void* const* d_in, const int* in_sizes, int n_in,
                              void* d_out, int out_size, void* d_ws, size_t ws_size,
                              hipStream_t stream) {
    (void)in_sizes; (void)n_in; (void)out_size; (void)ws_size;
    const int T = 4096, C = 768, H = 12;

    const float* x      = (const float*)d_in[0];
    const float* ln1_g  = (const float*)d_in[1];
    const float* ln1_b  = (const float*)d_in[2];
    const float* w_attn = (const float*)d_in[3];
    const float* b_attn = (const float*)d_in[4];
    const float* w_ao   = (const float*)d_in[5];
    const float* b_ao   = (const float*)d_in[6];
    const float* ln2_g  = (const float*)d_in[7];
    const float* ln2_b  = (const float*)d_in[8];
    const float* w_fc   = (const float*)d_in[9];
    const float* b_fc   = (const float*)d_in[10];
    const float* w_proj = (const float*)d_in[11];
    const float* b_proj = (const float*)d_in[12];
    float* out = (float*)d_out;

    char* wsp = (char*)d_ws;
    auto alloc = [&](size_t bytes) -> void* {
        void* p = (void*)wsp;
        wsp += (bytes + 255) & ~(size_t)255;
        return p;
    };
    const size_t TC2 = (size_t)T * C * 2;
    unsigned short* a1  = (unsigned short*)alloc(TC2);                 // ln1(x) bf16
    unsigned short* wtA = (unsigned short*)alloc((size_t)3 * C * C * 2);
    unsigned short* qw  = (unsigned short*)alloc(TC2);
    unsigned short* kw  = (unsigned short*)alloc(TC2);
    unsigned short* vtw = (unsigned short*)alloc(TC2);                 // V^T [C,T]
    unsigned short* yw  = (unsigned short*)alloc(TC2);                 // attn out bf16
    unsigned short* wtO = (unsigned short*)alloc((size_t)C * C * 2);
    float*          x1  = (float*)alloc((size_t)T * C * 4);            // residual stream
    unsigned short* a2  = (unsigned short*)alloc(TC2);                 // ln2(x1) bf16
    unsigned short* wtF = (unsigned short*)alloc((size_t)C * 4 * C * 2);
    unsigned short* hf  = (unsigned short*)alloc((size_t)T * 4 * C * 2);
    unsigned short* wtP = (unsigned short*)alloc((size_t)4 * C * C * 2);

    dim3 tb(32, 8);
    wcastT<<<dim3(3 * C / 32, C / 32), tb, 0, stream>>>(w_attn, wtA, C, 3 * C);
    wcastT<<<dim3(C / 32, C / 32),     tb, 0, stream>>>(w_ao,   wtO, C, C);
    wcastT<<<dim3(4 * C / 32, C / 32), tb, 0, stream>>>(w_fc,   wtF, C, 4 * C);
    wcastT<<<dim3(C / 32, 4 * C / 32), tb, 0, stream>>>(w_proj, wtP, 4 * C, C);

    ln_bf16<<<T, 256, 0, stream>>>(x, ln1_g, ln1_b, a1, C);

    // qkv = ln1(x) @ w_attn + b_attn, split into q/k and transposed v
    gemm_bf16_wmma<<<dim3(3 * C / 64, T / 128), 256, 0, stream>>>(
        a1, wtA, b_attn, T, 3 * C, C, 0, nullptr, nullptr, nullptr,
        qw, kw, vtw, C, T);

    attn_wmma<<<(H * (T / 16)) / 8, 256, 0, stream>>>(qw, kw, vtw, yw, T, C, H);

    // x1 = x + y @ w_ao + b_ao
    gemm_bf16_wmma<<<dim3(C / 64, T / 128), 256, 0, stream>>>(
        yw, wtO, b_ao, T, C, C, 1, x, x1, nullptr, nullptr, nullptr, nullptr, C, T);

    ln_bf16<<<T, 256, 0, stream>>>(x1, ln2_g, ln2_b, a2, C);

    // hf = gelu(ln2(x1) @ w_fc + b_fc)
    gemm_bf16_wmma<<<dim3(4 * C / 64, T / 128), 256, 0, stream>>>(
        a2, wtF, b_fc, T, 4 * C, C, 2, nullptr, nullptr, hf,
        nullptr, nullptr, nullptr, C, T);

    // out = x1 + hf @ w_proj + b_proj
    gemm_bf16_wmma<<<dim3(C / 64, T / 128), 256, 0, stream>>>(
        hf, wtP, b_proj, T, C, 4 * C, 1, x1, out, nullptr,
        nullptr, nullptr, nullptr, C, T);
}
